// MSSSM_33758442947270
// MI455X (gfx1250) — matrix operand order repaired and checked by reading, not compiled, not run on hardware
//
#include <hip/hip_runtime.h>
#include <hip/hip_bf16.h>
#include <cstdint>

#define DEV __device__ __forceinline__

typedef __attribute__((ext_vector_type(16))) __bf16 bf16x16;
typedef __attribute__((ext_vector_type(8)))  float  f32x8;

static constexpr int Lsz = 4096;   // H*W = 64*64
static constexpr int Bsz = 4;
static constexpr int Cch = 128;

// ---- WMMA fragment index helpers (CDNA5 ISA 7.12.2, wave32) ------------------
// 16-bit A (16x32): lane m = lane&15, group g = lane>>4.
// vector elem j -> K = 16g + j (contiguous, matches B)
DEV int wmma_a_k(int j, int g) {
    return 16 * g + j;
}
// 16-bit B (32x16): lane n = lane&15 ; elem j -> K = 16*(lane>>4) + j
// f32 C/D (16x16): elem r -> M = r + 8*(lane>>4), N = lane&15

DEV float sigmoidf_(float v) { return 1.f / (1.f + __expf(-v)); }
DEV float siluf_(float v)    { return v * sigmoidf_(v); }
DEV float softplusf_(float v){ return (v > 20.f) ? v : __logf(1.f + __expf(v)); }

// ============================================================================
// in_proj: xz[b,e,l] = sum_c in_w[e,c] * (x[b,coff1+c,l] (+ x[b,coff2+c,l]))
// GEMM M=128, K=32, N=B*L.  One wave per 16-col tile, 8 M-tiles.
// ============================================================================
__global__ __launch_bounds__(256) void k_in_proj(const float* __restrict__ x,
                                                 const float* __restrict__ w,
                                                 float* __restrict__ xz,
                                                 int coff1, int coff2)
{
    const int lane = threadIdx.x & 31;
    const int gw   = blockIdx.x * 8 + (threadIdx.x >> 5);  // ntile 0..1023
    const int n0   = gw << 4;
    const int b    = n0 >> 12;
    const int l0   = n0 & 4095;
    const int mr   = lane & 15;
    const int grp  = lane >> 4;
    const int col  = l0 + mr;

    const float* xb1 = x + ((size_t)b * Cch + coff1) * Lsz + col;
    const float* xb2 = x + ((size_t)b * Cch + ((coff2 >= 0) ? coff2 : coff1)) * Lsz + col;
    const bool add2 = (coff2 >= 0);

    bf16x16 bfr;
#pragma unroll
    for (int j = 0; j < 16; ++j) {
        int k = (grp << 4) + j;
        float v = xb1[(size_t)k * Lsz];
        if (add2) v += xb2[(size_t)k * Lsz];
        bfr[j] = (__bf16)v;
    }
#pragma unroll
    for (int mt = 0; mt < 8; ++mt) {
        bf16x16 afr;
#pragma unroll
        for (int j = 0; j < 16; ++j)
            afr[j] = (__bf16)w[(size_t)(mt * 16 + mr) * 32 + wmma_a_k(j, grp)];
        f32x8 acc = {};
        acc = __builtin_amdgcn_wmma_f32_16x16x32_bf16(false, afr, false, bfr,
                                                      (short)0, acc, false, false);
#pragma unroll
        for (int r = 0; r < 8; ++r) {
            int e = mt * 16 + r + (grp << 3);
            xz[((size_t)b * Cch + e) * Lsz + col] = acc[r];
        }
    }
}

// ============================================================================
// depthwise conv (3x3 or 7x7) + bias + SiLU on xi = xz channels [0,64)
// ============================================================================
__global__ __launch_bounds__(256) void k_dwconv(const float* __restrict__ xz,
                                                const float* __restrict__ cw,
                                                const float* __restrict__ cb,
                                                float* __restrict__ xc, int ks)
{
    int idx = blockIdx.x * 256 + threadIdx.x;           // b*64*L
    int l = idx & 4095, d = (idx >> 12) & 63, b = idx >> 18;
    int h = l >> 6, w = l & 63;
    int pad = (ks - 1) >> 1;
    const float* src = xz + ((size_t)b * Cch + d) * Lsz;
    float acc = cb[d];
    for (int kh = 0; kh < ks; ++kh) {
        int hh = h + kh - pad;
        if ((unsigned)hh >= 64u) continue;
        for (int kw = 0; kw < ks; ++kw) {
            int wc = w + kw - pad;
            if ((unsigned)wc >= 64u) continue;
            acc += src[hh * 64 + wc] * cw[(d * ks + kh) * ks + kw];
        }
    }
    xc[((size_t)b * 64 + d) * Lsz + l] = siluf_(acc);
}

// ============================================================================
// build the 4 directional sequences xs4[b,k,d,l]
// ============================================================================
__global__ __launch_bounds__(256) void k_build_xs4(const float* __restrict__ xc,
                                                   float* __restrict__ xs4)
{
    int idx = blockIdx.x * 256 + threadIdx.x;
    int l = idx & 4095, d = (idx >> 12) & 63, b = idx >> 18;
    float v = xc[((size_t)b * 64 + d) * Lsz + l];
    int h = l >> 6, w = l & 63;
    int lt = (w << 6) + h;                              // transposed traversal index
    size_t base = ((size_t)(b * 4) * 64 + d) * Lsz;
    xs4[base + l] = v;
    xs4[base + (size_t)64  * Lsz + lt] = v;
    xs4[base + (size_t)128 * Lsz + (4095 - l)]  = v;
    xs4[base + (size_t)192 * Lsz + (4095 - lt)] = v;
}

// ============================================================================
// x_dbl[b,k,c,l] = sum_d xp_w[k,c,d] * xs4[b,k,d,l]   GEMM M=Cp(pad16) K=64
// ============================================================================
__global__ __launch_bounds__(256) void k_xdbl(const float* __restrict__ xs4,
                                              const float* __restrict__ xpw,
                                              float* __restrict__ xdbl,
                                              int Cp, int mtiles)
{
    const int lane = threadIdx.x & 31;
    const int gw = blockIdx.x * 8 + (threadIdx.x >> 5); // 0..4095
    const int kdir = gw >> 10;
    const int nt = gw & 1023;
    const int n0 = nt << 4;
    const int b = n0 >> 12, l0 = n0 & 4095;
    const int mr = lane & 15, grp = lane >> 4, col = l0 + mr;

    const float* act = xs4 + (size_t)(b * 4 + kdir) * 64 * Lsz;
    const float* wb  = xpw + (size_t)kdir * Cp * 64;

    bf16x16 bfr[2];
#pragma unroll
    for (int kc = 0; kc < 2; ++kc)
#pragma unroll
        for (int j = 0; j < 16; ++j) {
            int ch = kc * 32 + (grp << 4) + j;
            bfr[kc][j] = (__bf16)act[(size_t)ch * Lsz + col];
        }

    for (int mt = 0; mt < mtiles; ++mt) {
        int m = mt * 16 + mr;
        f32x8 acc = {};
#pragma unroll
        for (int kc = 0; kc < 2; ++kc) {
            bf16x16 afr;
#pragma unroll
            for (int j = 0; j < 16; ++j) {
                int k = kc * 32 + wmma_a_k(j, grp);
                afr[j] = (__bf16)((m < Cp) ? wb[(size_t)m * 64 + k] : 0.f);
            }
            acc = __builtin_amdgcn_wmma_f32_16x16x32_bf16(false, afr, false, bfr[kc],
                                                          (short)0, acc, false, false);
        }
#pragma unroll
        for (int r = 0; r < 8; ++r) {
            int c = mt * 16 + r + (grp << 3);
            if (c < Cp)
                xdbl[((size_t)(b * 4 + kdir) * Cp + c) * Lsz + col] = acc[r];
        }
    }
}

// ============================================================================
// dts[b,k,d,l] = dt_w[k,d,0]*x_dbl[...,0,l] + dt_w[k,d,1]*x_dbl[...,1,l]
// ============================================================================
__global__ __launch_bounds__(256) void k_dts(const float* __restrict__ xdbl,
                                             const float* __restrict__ dtw,
                                             float* __restrict__ dts, int Cp)
{
    int idx = blockIdx.x * 256 + threadIdx.x;
    int l = idx & 4095, d = (idx >> 12) & 63, k = (idx >> 18) & 3, b = idx >> 20;
    const float* xd = xdbl + (size_t)(b * 4 + k) * Cp * Lsz;
    float r0 = xd[l], r1 = xd[(size_t)Lsz + l];
    dts[((size_t)((b * 4 + k) * 64) + d) * Lsz + l] =
        dtw[(k * 64 + d) * 2] * r0 + dtw[(k * 64 + d) * 2 + 1] * r1;
}

// ============================================================================
// selective scan: one lane per (b,dir,channel) = 1024 lanes, N states each
// ============================================================================
template <int N>
__global__ __launch_bounds__(256) void k_scan(const float* __restrict__ xs4,
                                              const float* __restrict__ dts,
                                              const float* __restrict__ xdbl,
                                              const float* __restrict__ A_log,
                                              const float* __restrict__ Dp,
                                              const float* __restrict__ dt_b,
                                              float* __restrict__ oy, int Cp)
{
    int idx = blockIdx.x * 256 + threadIdx.x;           // 0..1023
    int d = idx & 63, k = (idx >> 6) & 3, b = idx >> 8;
    float Ar[N], h[N];
#pragma unroll
    for (int n = 0; n < N; ++n) {
        Ar[n] = -__expf(A_log[(size_t)(k * 64 + d) * N + n]);
        h[n] = 0.f;
    }
    const float Dv = Dp[k * 64 + d];
    const float db = dt_b[k * 64 + d];
    const size_t row = (size_t)((b * 4 + k) * 64 + d) * Lsz;
    const float* u_row = xs4 + row;
    const float* dt_row = dts + row;
    const float* Bb = xdbl + ((size_t)(b * 4 + k) * Cp + 2) * Lsz;
    const float* Cb = Bb + (size_t)N * Lsz;
    float* o_row = oy + row;
    for (int l = 0; l < Lsz; ++l) {
        __builtin_prefetch(u_row + l + 64, 0, 0);
        __builtin_prefetch(dt_row + l + 64, 0, 0);
        float dt = softplusf_(dt_row[l] + db);
        float u = u_row[l];
        float du = dt * u;
        float acc = 0.f;
#pragma unroll
        for (int n = 0; n < N; ++n) {
            float hb = __expf(dt * Ar[n]) * h[n] + du * Bb[(size_t)n * Lsz + l];
            h[n] = hb;
            acc += hb * Cb[(size_t)n * Lsz + l];
        }
        o_row[l] = acc + u * Dv;
    }
}

// ============================================================================
// combine 4 directions back into spatial order
// ============================================================================
__global__ __launch_bounds__(256) void k_combine(const float* __restrict__ oy,
                                                 float* __restrict__ ysum)
{
    int idx = blockIdx.x * 256 + threadIdx.x;
    int l = idx & 4095, d = (idx >> 12) & 63, b = idx >> 18;
    int h = l >> 6, w = l & 63;
    int lt = (w << 6) + h;
    float v = oy[((size_t)(b * 4 + 0) * 64 + d) * Lsz + l]
            + oy[((size_t)(b * 4 + 2) * 64 + d) * Lsz + (4095 - l)]
            + oy[((size_t)(b * 4 + 1) * 64 + d) * Lsz + lt]
            + oy[((size_t)(b * 4 + 3) * 64 + d) * Lsz + (4095 - lt)];
    ysum[((size_t)b * 64 + d) * Lsz + l] = v;
}

// ============================================================================
// LayerNorm over channels (DI=64) + gating by silu(z)
// ============================================================================
__global__ __launch_bounds__(256) void k_ln_silu(const float* __restrict__ ysum,
                                                 const float* __restrict__ xz,
                                                 const float* __restrict__ lg,
                                                 const float* __restrict__ lb,
                                                 float* __restrict__ yln)
{
    int idx = blockIdx.x * 256 + threadIdx.x;           // b*L
    int l = idx & 4095, b = idx >> 12;
    const float* ys = ysum + (size_t)b * 64 * Lsz + l;
    float mu = 0.f;
    for (int d = 0; d < 64; ++d) mu += ys[(size_t)d * Lsz];
    mu *= (1.f / 64.f);
    float var = 0.f;
    for (int d = 0; d < 64; ++d) { float dv = ys[(size_t)d * Lsz] - mu; var += dv * dv; }
    float rs = rsqrtf(var * (1.f / 64.f) + 1e-5f);
    const float* zrow = xz + ((size_t)b * Cch + 64) * Lsz + l;
    float* orow = yln + (size_t)b * 64 * Lsz + l;
    for (int d = 0; d < 64; ++d) {
        float v = (ys[(size_t)d * Lsz] - mu) * rs * lg[d] + lb[d];
        orow[(size_t)d * Lsz] = v * siluf_(zrow[(size_t)d * Lsz]);
    }
}

// ============================================================================
// out_proj: s[b,c,l] = sum_d out_w[c,d]*yln[b,d,l]   GEMM M=32 K=64
// ============================================================================
__global__ __launch_bounds__(256) void k_out_proj(const float* __restrict__ yln,
                                                  const float* __restrict__ ow,
                                                  float* __restrict__ sbuf)
{
    const int lane = threadIdx.x & 31;
    const int gw = blockIdx.x * 8 + (threadIdx.x >> 5);
    const int n0 = gw << 4;
    const int b = n0 >> 12, l0 = n0 & 4095;
    const int mr = lane & 15, grp = lane >> 4, col = l0 + mr;
    const float* act = yln + (size_t)b * 64 * Lsz;

    bf16x16 bfr[2];
#pragma unroll
    for (int kc = 0; kc < 2; ++kc)
#pragma unroll
        for (int j = 0; j < 16; ++j) {
            int ch = kc * 32 + (grp << 4) + j;
            bfr[kc][j] = (__bf16)act[(size_t)ch * Lsz + col];
        }
#pragma unroll
    for (int mt = 0; mt < 2; ++mt) {
        f32x8 acc = {};
#pragma unroll
        for (int kc = 0; kc < 2; ++kc) {
            bf16x16 afr;
#pragma unroll
            for (int j = 0; j < 16; ++j)
                afr[j] = (__bf16)ow[(size_t)(mt * 16 + mr) * 64 + kc * 32 + wmma_a_k(j, grp)];
            acc = __builtin_amdgcn_wmma_f32_16x16x32_bf16(false, afr, false, bfr[kc],
                                                          (short)0, acc, false, false);
        }
#pragma unroll
        for (int r = 0; r < 8; ++r)
            sbuf[((size_t)b * 32 + mt * 16 + r + (grp << 3)) * Lsz + col] = acc[r];
    }
}

// ============================================================================
// pool over (H,W) of input slice(s); then bias from pw[:,32:64]
// ============================================================================
__global__ __launch_bounds__(128) void k_pool(const float* __restrict__ x,
                                              float* __restrict__ pool,
                                              int coff1, int coff2)
{
    int idx = threadIdx.x;                              // b*32
    int c = idx & 31, b = idx >> 5;
    const float* r1 = x + ((size_t)b * Cch + coff1 + c) * Lsz;
    const float* r2 = x + ((size_t)b * Cch + ((coff2 >= 0) ? coff2 : coff1) + c) * Lsz;
    bool add2 = coff2 >= 0;
    float s = 0.f;
    for (int l = 0; l < Lsz; ++l) { float v = r1[l]; if (add2) v += r2[l]; s += v; }
    pool[idx] = s * (1.f / 4096.f);
}

__global__ __launch_bounds__(128) void k_gbias(const float* __restrict__ pw,
                                               const float* __restrict__ pool,
                                               float* __restrict__ gbias)
{
    int idx = threadIdx.x;
    int co = idx & 31, b = idx >> 5;
    float a = 0.f;
    for (int ci = 0; ci < 32; ++ci) a += pw[co * 64 + 32 + ci] * pool[b * 32 + ci];
    gbias[idx] = a;
}

// ============================================================================
// group-conv 1x1 (pw[:,0:32] @ s) + gbias + BN + ReLU -> fused channels
// ============================================================================
__global__ __launch_bounds__(256) void k_gc_out(const float* __restrict__ sbuf,
                                                const float* __restrict__ pw,
                                                const float* __restrict__ gbias,
                                                const float* __restrict__ bg,
                                                const float* __restrict__ bb,
                                                const float* __restrict__ bm,
                                                const float* __restrict__ bv,
                                                float* __restrict__ fused, int choff)
{
    const int lane = threadIdx.x & 31;
    const int gw = blockIdx.x * 8 + (threadIdx.x >> 5);
    const int n0 = gw << 4;
    const int b = n0 >> 12, l0 = n0 & 4095;
    const int mr = lane & 15, grp = lane >> 4, col = l0 + mr;
    const float* act = sbuf + (size_t)b * 32 * Lsz;

    bf16x16 bfr;
#pragma unroll
    for (int j = 0; j < 16; ++j) {
        int ch = (grp << 4) + j;
        bfr[j] = (__bf16)act[(size_t)ch * Lsz + col];
    }
#pragma unroll
    for (int mt = 0; mt < 2; ++mt) {
        bf16x16 afr;
#pragma unroll
        for (int j = 0; j < 16; ++j)
            afr[j] = (__bf16)pw[(size_t)(mt * 16 + mr) * 64 + wmma_a_k(j, grp)];
        f32x8 acc = {};
        acc = __builtin_amdgcn_wmma_f32_16x16x32_bf16(false, afr, false, bfr,
                                                      (short)0, acc, false, false);
#pragma unroll
        for (int r = 0; r < 8; ++r) {
            int c = mt * 16 + r + (grp << 3);
            float s = bg[c] * rsqrtf(bv[c] + 1e-5f);
            float v = acc[r] + gbias[b * 32 + c];
            v = v * s + (bb[c] - bm[c] * s);
            fused[((size_t)b * Cch + choff + c) * Lsz + col] = fmaxf(v, 0.f);
        }
    }
}

// ============================================================================
// direct 3x3 conv on x1 + BN + ReLU -> fused[0:32]
// ============================================================================
__global__ __launch_bounds__(256) void k_conv1(const float* __restrict__ x,
                                               const float* __restrict__ w,
                                               const float* __restrict__ bg,
                                               const float* __restrict__ bb,
                                               const float* __restrict__ bm,
                                               const float* __restrict__ bv,
                                               float* __restrict__ fused)
{
    int idx = blockIdx.x * 256 + threadIdx.x;           // b*32*L
    int l = idx & 4095, co = (idx >> 12) & 31, b = idx >> 17;
    int h = l >> 6, ww = l & 63;
    const float* src = x + (size_t)b * Cch * Lsz;       // channels 0..31
    float acc = 0.f;
    for (int ci = 0; ci < 32; ++ci) {
        const float* sc = src + (size_t)ci * Lsz;
        const float* wk = w + (size_t)(co * 32 + ci) * 9;
        for (int kh = 0; kh < 3; ++kh) {
            int hh = h + kh - 1;
            if ((unsigned)hh >= 64u) continue;
            for (int kw = 0; kw < 3; ++kw) {
                int wc = ww + kw - 1;
                if ((unsigned)wc >= 64u) continue;
                acc += sc[hh * 64 + wc] * wk[kh * 3 + kw];
            }
        }
    }
    float s = bg[co] * rsqrtf(bv[co] + 1e-5f);
    acc = acc * s + (bb[co] - bm[co] * s);
    fused[((size_t)b * Cch + co) * Lsz + l] = fmaxf(acc, 0.f);
}

// ============================================================================
// dilated(12) 3x3 conv on (x3+x1) + ReLU -> fused[64:96]
// ============================================================================
__global__ __launch_bounds__(256) void k_conv3(const float* __restrict__ x,
                                               const float* __restrict__ w,
                                               float* __restrict__ fused)
{
    int idx = blockIdx.x * 256 + threadIdx.x;
    int l = idx & 4095, co = (idx >> 12) & 31, b = idx >> 17;
    int h = l >> 6, ww = l & 63;
    float acc = 0.f;
    for (int ci = 0; ci < 32; ++ci) {
        const float* s3 = x + ((size_t)b * Cch + 64 + ci) * Lsz;
        const float* s1 = x + ((size_t)b * Cch + ci) * Lsz;
        const float* wk = w + (size_t)(co * 32 + ci) * 9;
        for (int kh = 0; kh < 3; ++kh) {
            int hh = h + (kh - 1) * 12;
            if ((unsigned)hh >= 64u) continue;
            for (int kw = 0; kw < 3; ++kw) {
                int wc = ww + (kw - 1) * 12;
                if ((unsigned)wc >= 64u) continue;
                int p = hh * 64 + wc;
                acc += (s3[p] + s1[p]) * wk[kh * 3 + kw];
            }
        }
    }
    fused[((size_t)b * Cch + 64 + co) * Lsz + l] = fmaxf(acc, 0.f);
}

// ============================================================================
// per-channel means, gf = relu(gp_w @ xmean), pbias = proj_w[:,128:256] @ gf
// ============================================================================
__global__ __launch_bounds__(256) void k_cmean(const float* __restrict__ src,
                                               float* __restrict__ out)
{
    int idx = blockIdx.x * 256 + threadIdx.x;           // b*128
    const float* r = src + (size_t)idx * Lsz;
    float s = 0.f;
    for (int l = 0; l < Lsz; ++l) s += r[l];
    out[idx] = s * (1.f / 4096.f);
}

__global__ __launch_bounds__(256) void k_gf(const float* __restrict__ xm,
                                            const float* __restrict__ gpw,
                                            float* __restrict__ gf)
{
    int idx = blockIdx.x * 256 + threadIdx.x;
    int c = idx & 127, b = idx >> 7;
    float a = 0.f;
    for (int i = 0; i < 128; ++i) a += gpw[c * 128 + i] * xm[b * 128 + i];
    gf[idx] = fmaxf(a, 0.f);
}

__global__ __launch_bounds__(256) void k_pbias(const float* __restrict__ gf,
                                               const float* __restrict__ pw,
                                               float* __restrict__ pbias)
{
    int idx = blockIdx.x * 256 + threadIdx.x;
    int co = idx & 127, b = idx >> 7;
    float a = 0.f;
    for (int ci = 0; ci < 128; ++ci) a += pw[co * 256 + 128 + ci] * gf[b * 128 + ci];
    pbias[idx] = a;
}

// ============================================================================
// proj: y = x + relu(BN(proj_w[:,0:128] @ fused + pbias))  GEMM M=128 K=128
// Activation panel (128x16 fp32, reused by all 8 M-tiles) is staged into LDS
// with GLOBAL_LOAD_ASYNC_TO_LDS_B128 (ASYNCcnt), then consumed via ds_load.
// 4 waves/block, one private 8KB tile per wave -> 32KB LDS, no barrier needed
// (each wave waits only on its own ASYNCcnt).
// ============================================================================
__global__ __launch_bounds__(128) void k_proj(const float* __restrict__ fused,
                                              const float* __restrict__ pw,
                                              const float* __restrict__ pbias,
                                              const float* __restrict__ bg,
                                              const float* __restrict__ bb,
                                              const float* __restrict__ bm,
                                              const float* __restrict__ bv,
                                              const float* __restrict__ x,
                                              float* __restrict__ ybuf)
{
    __shared__ __align__(16) float tile[4][128][16];
    const int lane = threadIdx.x & 31;
    const int wv = threadIdx.x >> 5;
    const int gw = blockIdx.x * 4 + wv;                 // ntile 0..1023
    const int n0 = gw << 4;
    const int b = n0 >> 12, l0 = n0 & 4095;
    const int mr = lane & 15, grp = lane >> 4, col = l0 + mr;
    const float* act = fused + (size_t)b * Cch * Lsz + l0;

    // async-stage 128x16 fp32 panel: 512 x 16B chunks spread over 32 lanes
#pragma unroll
    for (int it = 0; it < 16; ++it) {
        int chunk = it * 32 + lane;                     // 0..511
        int ch = chunk >> 2;
        int part = (chunk & 3) << 2;                    // float offset in row
        const float* g = act + (size_t)ch * Lsz + part;
        unsigned int ldsoff = (unsigned int)(size_t)&tile[wv][ch][part];
        asm volatile("global_load_async_to_lds_b128 %0, %1, off"
                     :: "v"(ldsoff), "v"(g) : "memory");
    }
    asm volatile("s_wait_asynccnt 0" ::: "memory");

    bf16x16 bfr[4];
#pragma unroll
    for (int kc = 0; kc < 4; ++kc)
#pragma unroll
        for (int j = 0; j < 16; ++j)
            bfr[kc][j] = (__bf16)tile[wv][kc * 32 + (grp << 4) + j][mr];

#pragma unroll
    for (int mt = 0; mt < 8; ++mt) {
        f32x8 acc = {};
#pragma unroll
        for (int kc = 0; kc < 4; ++kc) {
            bf16x16 afr;
#pragma unroll
            for (int j = 0; j < 16; ++j)
                afr[j] = (__bf16)pw[(size_t)(mt * 16 + mr) * 256 + kc * 32 + wmma_a_k(j, grp)];
            acc = __builtin_amdgcn_wmma_f32_16x16x32_bf16(false, afr, false, bfr[kc],
                                                          (short)0, acc, false, false);
        }
#pragma unroll
        for (int r = 0; r < 8; ++r) {
            int c = mt * 16 + r + (grp << 3);
            float s = bg[c] * rsqrtf(bv[c] + 1e-5f);
            float v = acc[r] + pbias[b * 128 + c];
            v = v * s + (bb[c] - bm[c] * s);
            size_t o = ((size_t)b * Cch + c) * Lsz + col;
            ybuf[o] = fmaxf(v, 0.f) + x[o];
        }
    }
}

// ============================================================================
// channel attention (LDS + workgroup barrier), spatial attention, final blend
// ============================================================================
__global__ __launch_bounds__(128) void k_ca(const float* __restrict__ ym,
                                            const float* __restrict__ w1,
                                            const float* __restrict__ b1,
                                            const float* __restrict__ w2,
                                            const float* __restrict__ b2,
                                            float* __restrict__ ca)
{
    __shared__ float mid[8];
    int b = blockIdx.x, c = threadIdx.x;
    if (c < 8) {
        float a = b1[c];
        for (int i = 0; i < 128; ++i) a += w1[c * 128 + i] * ym[b * 128 + i];
        mid[c] = fmaxf(a, 0.f);
    }
    __syncthreads();
    float a = b2[c];
#pragma unroll
    for (int j = 0; j < 8; ++j) a += w2[c * 8 + j] * mid[j];
    ca[b * 128 + c] = sigmoidf_(a);
}

__global__ __launch_bounds__(256) void k_sa(const float* __restrict__ y,
                                            const float* __restrict__ w,
                                            float* __restrict__ sa)
{
    int idx = blockIdx.x * 256 + threadIdx.x;           // b*L
    int l = idx & 4095, b = idx >> 12;
    int h = l >> 6, ww = l & 63;
    float acc = 0.f;
    for (int c = 0; c < 128; ++c) {
        const float* sc = y + ((size_t)b * Cch + c) * Lsz;
        const float* wk = w + c * 49;
        for (int kh = 0; kh < 7; ++kh) {
            int hh = h + kh - 3;
            if ((unsigned)hh >= 64u) continue;
            for (int kw = 0; kw < 7; ++kw) {
                int wc = ww + kw - 3;
                if ((unsigned)wc >= 64u) continue;
                acc += sc[hh * 64 + wc] * wk[kh * 7 + kw];
            }
        }
    }
    sa[idx] = sigmoidf_(acc);
}

__global__ __launch_bounds__(256) void k_final(const float* __restrict__ y,
                                               const float* __restrict__ ca,
                                               const float* __restrict__ sa,
                                               float* __restrict__ out)
{
    int idx = blockIdx.x * 256 + threadIdx.x;           // b*128*L
    int l = idx & 4095, c = (idx >> 12) & 127, b = idx >> 19;
    float v = y[idx];
    out[idx] = v * (ca[b * 128 + c] * sa[b * 4096 + l]) + v;
}

// ============================================================================
extern "C" void kernel_launch(void* const* d_in, const int* /*in_sizes*/, int /*n_in*/,
                              void* d_out, int /*out_size*/, void* d_ws, size_t /*ws_size*/,
                              hipStream_t stream)
{
    int i = 0;
    const float* x          = (const float*)d_in[i++]; // 0
    const float* conv1_w    = (const float*)d_in[i++];
    const float* c1g = (const float*)d_in[i++]; const float* c1b = (const float*)d_in[i++];
    const float* c1m = (const float*)d_in[i++]; const float* c1v = (const float*)d_in[i++];
    const float* s2_inw = (const float*)d_in[i++]; const float* s2_cw = (const float*)d_in[i++];
    const float* s2_cb  = (const float*)d_in[i++]; const float* s2_xp = (const float*)d_in[i++];
    const float* s2_dtw = (const float*)d_in[i++]; const float* s2_dtb = (const float*)d_in[i++];
    const float* s2_Al  = (const float*)d_in[i++]; const float* s2_D  = (const float*)d_in[i++];
    const float* s2_lg  = (const float*)d_in[i++]; const float* s2_lb = (const float*)d_in[i++];
    const float* s2_ow  = (const float*)d_in[i++];
    const float* g2_pw  = (const float*)d_in[i++];
    const float* g2g = (const float*)d_in[i++]; const float* g2b = (const float*)d_in[i++];
    const float* g2m = (const float*)d_in[i++]; const float* g2v = (const float*)d_in[i++];
    const float* conv3_w = (const float*)d_in[i++];
    const float* s4_inw = (const float*)d_in[i++]; const float* s4_cw = (const float*)d_in[i++];
    const float* s4_cb  = (const float*)d_in[i++]; const float* s4_xp = (const float*)d_in[i++];
    const float* s4_dtw = (const float*)d_in[i++]; const float* s4_dtb = (const float*)d_in[i++];
    const float* s4_Al  = (const float*)d_in[i++]; const float* s4_D  = (const float*)d_in[i++];
    const float* s4_lg  = (const float*)d_in[i++]; const float* s4_lb = (const float*)d_in[i++];
    const float* s4_ow  = (const float*)d_in[i++];
    const float* g4_pw  = (const float*)d_in[i++];
    const float* g4g = (const float*)d_in[i++]; const float* g4b = (const float*)d_in[i++];
    const float* g4m = (const float*)d_in[i++]; const float* g4v = (const float*)d_in[i++];
    const float* gp_w  = (const float*)d_in[i++];
    const float* proj_w = (const float*)d_in[i++];
    const float* pg = (const float*)d_in[i++]; const float* pb = (const float*)d_in[i++];
    const float* pm = (const float*)d_in[i++]; const float* pv = (const float*)d_in[i++];
    const float* ca_w1 = (const float*)d_in[i++]; const float* ca_b1 = (const float*)d_in[i++];
    const float* ca_w2 = (const float*)d_in[i++]; const float* ca_b2 = (const float*)d_in[i++];
    const float* sa_w  = (const float*)d_in[i++]; // 49

    float* ws = (float*)d_ws;
    size_t off = 0;
    auto alloc = [&](size_t n) { float* p = ws + off; off += n; return p; };
    float* fused = alloc((size_t)Bsz * 128 * Lsz);
    float* ybuf  = alloc((size_t)Bsz * 128 * Lsz);
    float* xz    = alloc((size_t)Bsz * 128 * Lsz);
    float* xc    = alloc((size_t)Bsz * 64 * Lsz);
    float* xs4   = alloc((size_t)Bsz * 4 * 64 * Lsz);
    float* xdbl  = alloc((size_t)Bsz * 4 * 50 * Lsz);
    float* dts   = alloc((size_t)Bsz * 4 * 64 * Lsz);
    float* oy    = alloc((size_t)Bsz * 4 * 64 * Lsz);
    float* ysum  = alloc((size_t)Bsz * 64 * Lsz);
    float* yln   = alloc((size_t)Bsz * 64 * Lsz);
    float* sbuf  = alloc((size_t)Bsz * 32 * Lsz);
    float* pool  = alloc(Bsz * 32);
    float* gbias = alloc(Bsz * 32);
    float* xmean = alloc(Bsz * 128);
    float* gf    = alloc(Bsz * 128);
    float* pbias = alloc(Bsz * 128);
    float* ymean = alloc(Bsz * 128);
    float* cab   = alloc(Bsz * 128);
    float* sab   = alloc((size_t)Bsz * Lsz);

    auto run_ss = [&](const float* in_w, const float* cw, const float* cb,
                      const float* xp_w, const float* dt_w, const float* dt_b,
                      const float* A_log, const float* Dp,
                      const float* ln_g, const float* ln_b, const float* out_w,
                      const float* pw, const float* bg, const float* bb,
                      const float* bm, const float* bv,
                      int ks, int Nst, int coff1, int coff2, int choff) {
        int Cp = 2 + 2 * Nst;
        int mtiles = (Cp + 15) / 16;
        k_in_proj<<<128, 256, 0, stream>>>(x, in_w, xz, coff1, coff2);
        k_dwconv<<<4096, 256, 0, stream>>>(xz, cw, cb, xc, ks);
        k_build_xs4<<<4096, 256, 0, stream>>>(xc, xs4);
        k_xdbl<<<512, 256, 0, stream>>>(xs4, xp_w, xdbl, Cp, mtiles);
        k_dts<<<16384, 256, 0, stream>>>(xdbl, dt_w, dts, Cp);
        if (Nst == 8)
            k_scan<8><<<4, 256, 0, stream>>>(xs4, dts, xdbl, A_log, Dp, dt_b, oy, Cp);
        else
            k_scan<24><<<4, 256, 0, stream>>>(xs4, dts, xdbl, A_log, Dp, dt_b, oy, Cp);
        k_combine<<<4096, 256, 0, stream>>>(oy, ysum);
        k_ln_silu<<<64, 256, 0, stream>>>(ysum, xz, ln_g, ln_b, yln);
        k_out_proj<<<128, 256, 0, stream>>>(yln, out_w, sbuf);
        k_pool<<<1, 128, 0, stream>>>(x, pool, coff1, coff2);
        k_gbias<<<1, 128, 0, stream>>>(pw, pool, gbias);
        k_gc_out<<<128, 256, 0, stream>>>(sbuf, pw, gbias, bg, bb, bm, bv, fused, choff);
    };

    // branch 2: SS2D on x2 (channels 32..63), N=8, ks=3 -> fused[32:64]
    run_ss(s2_inw, s2_cw, s2_cb, s2_xp, s2_dtw, s2_dtb, s2_Al, s2_D,
           s2_lg, s2_lb, s2_ow, g2_pw, g2g, g2b, g2m, g2v, 3, 8, 32, -1, 32);
    // branch 4: SS2D on x4+x2 (channels 96 + 32), N=24, ks=7 -> fused[96:128]
    run_ss(s4_inw, s4_cw, s4_cb, s4_xp, s4_dtw, s4_dtb, s4_Al, s4_D,
           s4_lg, s4_lb, s4_ow, g4_pw, g4g, g4b, g4m, g4v, 7, 24, 96, 32, 96);
    // branch 1 and 3
    k_conv1<<<2048, 256, 0, stream>>>(x, conv1_w, c1g, c1b, c1m, c1v, fused);
    k_conv3<<<2048, 256, 0, stream>>>(x, conv3_w, fused);
    // global feature + proj
    k_cmean<<<2, 256, 0, stream>>>(x, xmean);
    k_gf<<<2, 256, 0, stream>>>(xmean, gp_w, gf);
    k_pbias<<<2, 256, 0, stream>>>(gf, proj_w, pbias);
    k_proj<<<256, 128, 0, stream>>>(fused, proj_w, pbias, pg, pb, pm, pv, x, ybuf);
    // attention + final
    k_cmean<<<2, 256, 0, stream>>>(ybuf, ymean);
    k_ca<<<4, 128, 0, stream>>>(ymean, ca_w1, ca_b1, ca_w2, ca_b2, cab);
    k_sa<<<64, 256, 0, stream>>>(ybuf, sa_w, sab);
    k_final<<<8192, 256, 0, stream>>>(ybuf, cab, sab, (float*)d_out);
}
